// GATLayer_48859547959813
// MI455X (gfx1250) — matrix operand order repaired and checked by reading, not compiled, not run on hardware
//
#include <hip/hip_runtime.h>

// Problem dims (fixed by reference)
#define B_ 32
#define C_ 512
#define N_ 1024
#define O_ 512

typedef __attribute__((ext_vector_type(16))) _Float16 v16h;
typedef __attribute__((ext_vector_type(8)))  float    v8f;

#if defined(__has_builtin)
#if __has_builtin(__builtin_amdgcn_global_load_async_to_lds_b128)
#define ASYNC_CP 1
#endif
#endif

// Typed address-space pointers for the async-copy builtin:
// param0: int4 in global (AS1), param1: int4 in LDS (AS3), then (imm offset, imm cpol)
typedef __attribute__((ext_vector_type(4))) int i32x4;
typedef __attribute__((address_space(1))) i32x4* gvecp;
typedef __attribute__((address_space(3))) i32x4* lvecp;

#define GAS1(p) ((gvecp)(void*)(p))
#define LAS3(p) ((lvecp)(void*)(p))

__device__ __forceinline__ void wait_async0() {
#if defined(__has_builtin) && __has_builtin(__builtin_amdgcn_s_wait_asynccnt)
    __builtin_amdgcn_s_wait_asynccnt(0);
#else
    asm volatile("s_wait_asynccnt 0x0" ::: "memory");
#endif
}

// ---------------------------------------------------------------------------
// Kernel 0: convert wm (O x C, f32) -> f16
// ---------------------------------------------------------------------------
__global__ void k_wm_cvt(const float* __restrict__ wm, _Float16* __restrict__ wmh) {
    int i = blockIdx.x * blockDim.x + threadIdx.x;
    if (i < O_ * C_) wmh[i] = (_Float16)wm[i];
}

// ---------------------------------------------------------------------------
// Kernel 1: q[b,n] = sum_c wq[c]*x[b,c,n]; k likewise; xh = f16(x)
// ---------------------------------------------------------------------------
__global__ void k_qk_xh(const float* __restrict__ x, const float* __restrict__ wq,
                        const float* __restrict__ wk, float* __restrict__ qv,
                        float* __restrict__ kv, _Float16* __restrict__ xh) {
    __shared__ float swq[C_];
    __shared__ float swk[C_];
    for (int c = threadIdx.x; c < C_; c += blockDim.x) { swq[c] = wq[c]; swk[c] = wk[c]; }
    __syncthreads();

    int idx = blockIdx.x * blockDim.x + threadIdx.x;   // over B*N
    int b = idx / N_, n = idx % N_;
    const float* xp = x + (size_t)b * C_ * N_ + n;
    _Float16* xhp = xh + (size_t)b * C_ * N_ + n;
    float qa = 0.f, ka = 0.f;
#pragma unroll 4
    for (int c = 0; c < C_; ++c) {
        float v = xp[(size_t)c * N_];
        qa += swq[c] * v;
        ka += swk[c] * v;
        xhp[(size_t)c * N_] = (_Float16)v;
    }
    qv[idx] = qa;
    kv[idx] = ka;
}

// ---------------------------------------------------------------------------
// Kernel 2: att[b,m,n] = softmax_n( leakyrelu(q[b,n] + k[b,m]) ), stored f16.
// ---------------------------------------------------------------------------
__global__ void k_softmax(const float* __restrict__ qv, const float* __restrict__ kv,
                          _Float16* __restrict__ att) {
    const int b = blockIdx.y;
    const int m = blockIdx.x;
    const int t = threadIdx.x;
    const int lane = t & 31, wv = t >> 5;

    const float km = kv[b * N_ + m];
    const float* qp = qv + (size_t)b * N_;

    float s[4];
    float mx = -1e30f;
#pragma unroll
    for (int i = 0; i < 4; ++i) {
        float v = qp[t + i * 256] + km;
        v = v > 0.f ? v : 0.01f * v;      // LeakyReLU(0.01)
        s[i] = v;
        mx = fmaxf(mx, v);
    }

    __shared__ float redmax[8];
    for (int off = 16; off; off >>= 1) mx = fmaxf(mx, __shfl_xor(mx, off, 32));
    if (lane == 0) redmax[wv] = mx;
    __syncthreads();
    mx = redmax[0];
#pragma unroll
    for (int w = 1; w < 8; ++w) mx = fmaxf(mx, redmax[w]);

    float sum = 0.f;
#pragma unroll
    for (int i = 0; i < 4; ++i) { s[i] = __expf(s[i] - mx); sum += s[i]; }
    for (int off = 16; off; off >>= 1) sum += __shfl_xor(sum, off, 32);
    __shared__ float redsum[8];
    if (lane == 0) redsum[wv] = sum;
    __syncthreads();
    sum = 0.f;
#pragma unroll
    for (int w = 0; w < 8; ++w) sum += redsum[w];
    const float inv = 1.0f / sum;

    _Float16* ap = att + ((size_t)b * N_ + m) * N_;
#pragma unroll
    for (int i = 0; i < 4; ++i) ap[t + i * 256] = (_Float16)(s[i] * inv);
}

// ---------------------------------------------------------------------------
// WMMA fragment loaders (16-bit A 16x32 / B 32x16; this code uses contiguous k = 16*hi + j for both)
// ---------------------------------------------------------------------------
__device__ __forceinline__ v16h load_a_frag(const _Float16* __restrict__ rowp, int hi) {
    v16h r;
#pragma unroll
    for (int j = 0; j < 16; ++j) r[j] = rowp[hi * 16 + j];
    return r;
}
__device__ __forceinline__ v16h load_b_frag(const _Float16* __restrict__ rowp, int hi) {
    v16h r;
#pragma unroll
    for (int j = 0; j < 16; ++j) r[j] = rowp[hi * 16 + j];
    return r;
}

// ---------------------------------------------------------------------------
// Shared GEMM core: block tile 64(M) x 128(N), K-step 32, 128 threads = 4 waves
// in 2(M)x2(N); each wave computes 32x64 -> 2 A-frags x 4 B-frags = 8 WMMA/stage.
// Double-buffered LDS: async DMA for stage s+1 overlaps the WMMAs of stage s.
// ---------------------------------------------------------------------------
template <int KDIM>
__device__ __forceinline__ void gemm_core(const _Float16* __restrict__ Ag, int strideKA,
                                          const _Float16* __restrict__ Bg, int strideKB,
                                          _Float16* As0, _Float16* As1,
                                          _Float16* Bs0, _Float16* Bs1,
                                          int t, int wm2, int wn2, int hi, int lo,
                                          v8f acc[2][4]) {
    const int rowA = t >> 1;           // 0..63
    const int colh = (t & 1) * 16;     // 0 or 16 halfs

#ifdef ASYNC_CP
    auto issue = [&](int ks, _Float16* As, _Float16* Bs) {
        const _Float16* ga = Ag + (size_t)rowA * strideKA + ks + colh;
        _Float16* la = As + rowA * 32 + colh;
        __builtin_amdgcn_global_load_async_to_lds_b128(GAS1(ga),     LAS3(la),     0, 0);
        __builtin_amdgcn_global_load_async_to_lds_b128(GAS1(ga + 8), LAS3(la + 8), 0, 0);
        const _Float16* gb = Bg + (size_t)t * strideKB + ks;
        _Float16* lb = Bs + t * 32;
#pragma unroll
        for (int u = 0; u < 4; ++u)
            __builtin_amdgcn_global_load_async_to_lds_b128(GAS1(gb + u * 8), LAS3(lb + u * 8), 0, 0);
    };

    issue(0, As0, Bs0);    // prologue: stage 0 into buffer 0

    for (int ks = 0; ks < KDIM; ks += 32) {
        const int cur = (ks >> 5) & 1;
        _Float16* Asc = cur ? As1 : As0;
        _Float16* Bsc = cur ? Bs1 : Bs0;
        _Float16* Asn = cur ? As0 : As1;
        _Float16* Bsn = cur ? Bs0 : Bs1;

        wait_async0();     // my stage-s tile (only outstanding group) has landed
        __syncthreads();   // all waves have it; all waves finished compute(s-1)
        if (ks + 32 < KDIM)
            issue(ks + 32, Asn, Bsn);   // DMA for s+1 overlaps WMMAs below
#else
    for (int ks = 0; ks < KDIM; ks += 32) {
        _Float16* Asc = As0;
        _Float16* Bsc = Bs0;
        const uint4* ga = (const uint4*)(Ag + (size_t)rowA * strideKA + ks + colh);
        uint4 av0 = ga[0], av1 = ga[1];
        const uint4* gb = (const uint4*)(Bg + (size_t)t * strideKB + ks);
        uint4 bv0 = gb[0], bv1 = gb[1], bv2 = gb[2], bv3 = gb[3];
        if (ks + 32 < KDIM) {
            __builtin_prefetch(Ag + (size_t)rowA * strideKA + ks + 32 + colh, 0, 1);
            __builtin_prefetch(Bg + (size_t)t * strideKB + ks + 32, 0, 1);
        }
        __syncthreads();
        *(uint4*)(Asc + rowA * 32 + colh)     = av0;
        *(uint4*)(Asc + rowA * 32 + colh + 8) = av1;
        uint4* lb = (uint4*)(Bsc + t * 32);
        lb[0] = bv0; lb[1] = bv1; lb[2] = bv2; lb[3] = bv3;
        __syncthreads();
#endif
        v16h afr[2], bfr[4];
#pragma unroll
        for (int i = 0; i < 2; ++i)
            afr[i] = load_a_frag(Asc + (wm2 * 32 + i * 16 + lo) * 32, hi);
#pragma unroll
        for (int j = 0; j < 4; ++j)
            bfr[j] = load_b_frag(Bsc + (wn2 * 64 + j * 16 + lo) * 32, hi);
#pragma unroll
        for (int i = 0; i < 2; ++i)
#pragma unroll
            for (int j = 0; j < 4; ++j)
                acc[i][j] = __builtin_amdgcn_wmma_f32_16x16x32_f16(
                    false, afr[i], false, bfr[j], (short)0, acc[i][j], false, false);
    }
}

// ---------------------------------------------------------------------------
// Kernel 3: aggT[b][m][c] = sum_n xh[b][c][n] * att[b][m][n]   (f16 WMMA GEMM)
// ---------------------------------------------------------------------------
__global__ __launch_bounds__(128) void k_gemm_agg(const _Float16* __restrict__ xh,
                                                  const _Float16* __restrict__ att,
                                                  _Float16* __restrict__ aggT) {
    __shared__ __align__(16) _Float16 As[2][64 * 32];
    __shared__ __align__(16) _Float16 Bs[2][128 * 32];

    const int b  = blockIdx.z;
    const int c0 = blockIdx.y * 64;       // M dim (c)
    const int m0 = blockIdx.x * 128;      // N dim (m)
    const int t = threadIdx.x;
    const int lane = t & 31, wv = t >> 5;
    const int wm2 = wv >> 1, wn2 = wv & 1;
    const int hi = lane >> 4, lo = lane & 15;

    const _Float16* Ag = xh  + ((size_t)b * C_ + c0) * N_;   // row stride N_
    const _Float16* Bg = att + ((size_t)b * N_ + m0) * N_;   // row stride N_

    v8f acc[2][4] = {};
    gemm_core<N_>(Ag, N_, Bg, N_, As[0], As[1], Bs[0], Bs[1], t, wm2, wn2, hi, lo, acc);

    // Epilogue: D(M=c, N=m) -> aggT[b][m][c]; per-lane 8 contiguous c values.
#pragma unroll
    for (int i = 0; i < 2; ++i) {
#pragma unroll
        for (int j = 0; j < 4; ++j) {
            const int mg = m0 + wn2 * 64 + j * 16 + lo;
            const int cg = c0 + wm2 * 32 + i * 16 + hi * 8;
            __align__(16) _Float16 h[8];
#pragma unroll
            for (int r = 0; r < 8; ++r) h[r] = (_Float16)acc[i][j][r];
            *(uint4*)(aggT + ((size_t)b * N_ + mg) * C_ + cg) = *(const uint4*)h;
        }
    }
}

// ---------------------------------------------------------------------------
// Kernel 4: out[b][o][m] = relu( sum_c wmh[o][c]*aggT[b][m][c] + bm[o] ), f32
// ---------------------------------------------------------------------------
__global__ __launch_bounds__(128) void k_gemm_out(const _Float16* __restrict__ wmh,
                                                  const _Float16* __restrict__ aggT,
                                                  const float* __restrict__ bm,
                                                  float* __restrict__ out) {
    __shared__ __align__(16) _Float16 As[2][64 * 32];
    __shared__ __align__(16) _Float16 Bs[2][128 * 32];

    const int b  = blockIdx.z;
    const int o0 = blockIdx.y * 64;       // M dim (o)
    const int m0 = blockIdx.x * 128;      // N dim (m)
    const int t = threadIdx.x;
    const int lane = t & 31, wv = t >> 5;
    const int wm2 = wv >> 1, wn2 = wv & 1;
    const int hi = lane >> 4, lo = lane & 15;

    const _Float16* Ag = wmh  + (size_t)o0 * C_;              // row stride C_
    const _Float16* Bg = aggT + ((size_t)b * N_ + m0) * C_;   // row stride C_

    v8f acc[2][4] = {};
    gemm_core<C_>(Ag, C_, Bg, C_, As[0], As[1], Bs[0], Bs[1], t, wm2, wn2, hi, lo, acc);

    // Epilogue: bias + ReLU, f32 stores (stride N_ across the 8 rows per lane)
#pragma unroll
    for (int i = 0; i < 2; ++i) {
#pragma unroll
        for (int j = 0; j < 4; ++j) {
            const int mg  = m0 + wn2 * 64 + j * 16 + lo;
            const int og0 = o0 + wm2 * 32 + i * 16 + hi * 8;
#pragma unroll
            for (int r = 0; r < 8; ++r) {
                float v = acc[i][j][r] + bm[og0 + r];
                out[((size_t)b * O_ + og0 + r) * N_ + mg] = v > 0.f ? v : 0.f;
            }
        }
    }
}

// ---------------------------------------------------------------------------
// Launch
// ---------------------------------------------------------------------------
extern "C" void kernel_launch(void* const* d_in, const int* in_sizes, int n_in,
                              void* d_out, int out_size, void* d_ws, size_t ws_size,
                              hipStream_t stream) {
    const float* x  = (const float*)d_in[0];
    const float* wq = (const float*)d_in[1];
    const float* wk = (const float*)d_in[2];
    const float* wm = (const float*)d_in[3];
    const float* bm = (const float*)d_in[4];
    float* out = (float*)d_out;

    char* ws = (char*)d_ws;
    const size_t qO   = 0;                                     // B*N f32
    const size_t kO   = qO   + (size_t)B_ * N_ * 4;            // B*N f32
    const size_t xhO  = kO   + (size_t)B_ * N_ * 4;            // B*C*N f16
    const size_t wmhO = xhO  + (size_t)B_ * C_ * N_ * 2;       // O*C f16
    const size_t attO = wmhO + (size_t)O_ * C_ * 2;            // B*N*N f16
    const size_t aggO = attO + (size_t)B_ * N_ * N_ * 2;       // B*N*C f16

    float*    qv   = (float*)(ws + qO);
    float*    kv   = (float*)(ws + kO);
    _Float16* xh   = (_Float16*)(ws + xhO);
    _Float16* wmh  = (_Float16*)(ws + wmhO);
    _Float16* att  = (_Float16*)(ws + attO);
    _Float16* aggT = (_Float16*)(ws + aggO);

    k_wm_cvt<<<dim3((O_ * C_ + 255) / 256), 256, 0, stream>>>(wm, wmh);
    k_qk_xh<<<dim3(B_ * N_ / 256), 256, 0, stream>>>(x, wq, wk, qv, kv, xh);
    k_softmax<<<dim3(N_, B_), 256, 0, stream>>>(qv, kv, att);
    k_gemm_agg<<<dim3(N_ / 128, C_ / 64, B_), 128, 0, stream>>>(xh, att, aggT);
    k_gemm_out<<<dim3(N_ / 128, O_ / 64, B_), 128, 0, stream>>>(wmh, aggT, bm, out);
}
